// TransIBRNet_53188874993898
// MI455X (gfx1250) — matrix operand order repaired and checked by reading, not compiled, not run on hardware
//
#include <hip/hip_runtime.h>
#include <hip/hip_bf16.h>

// ---------------------------------------------------------------------------
// Types for CDNA5 WMMA
// ---------------------------------------------------------------------------
typedef __attribute__((ext_vector_type(16))) _Float16 v16h;
typedef __attribute__((ext_vector_type(8)))  float    v8f;
typedef __attribute__((ext_vector_type(8)))  _Float16 h8v;
typedef __attribute__((ext_vector_type(2)))  _Float16 h2v;

#if __has_builtin(__builtin_amdgcn_tensor_load_to_lds) && \
    __has_builtin(__builtin_amdgcn_s_wait_tensorcnt)
#define HAVE_TDM 1
typedef unsigned int u32x4 __attribute__((ext_vector_type(4)));
typedef int          i32x8 __attribute__((ext_vector_type(8)));
typedef int          i32x4 __attribute__((ext_vector_type(4)));
#else
#define HAVE_TDM 0
#endif

#define DIMC   128
#define HEADS  4
#define DH     32
#define NRAY   1024
#define SS     64
#define VV     10
#define NS     (NRAY * SS)          // 65536
#define NSV    (NS * VV)            // 655360
#define FEATC  35

#define BM 128
#define BN 64
#define KC 32
#define LDA_S 40    // padded LDS stride for chunked A staging
#define LDB_S 66
#define AF_S  136   // padded LDS stride for full 128-wide A tile (TDM pad: 4dw / 64dw)

// ---------------------------------------------------------------------------
// Generic WMMA GEMM (fully padded / branch-free staging):
//   out[m,n] = act( A[m,:K] . W[n,:K] + bias[n] + resid[m,n] )
// REQUIREMENTS (host-guaranteed): M%128==0, K%32==0, 16B-aligned rows,
// Nt%16==0. Block: 256 thr = 8 waves; tile 128(M) x 64(N); K chunked by 32.
// ---------------------------------------------------------------------------
__global__ __launch_bounds__(256) void gemm_f16_wmma(
    const _Float16* __restrict__ A, int lda,
    const _Float16* __restrict__ W, int ldw,
    const float* __restrict__ bias,
    const float* __restrict__ resid,
    float* __restrict__ outF,
    _Float16* __restrict__ outH,
    int M, int Nt, int K, int ldo, int relu)
{
    __shared__ __align__(16) _Float16 As[BM * LDA_S];
    __shared__ __align__(16) _Float16 Bs[KC * LDB_S];

    const int tid  = threadIdx.x;
    const int lane = tid & 31;
    const int wv   = tid >> 5;
    const int wm   = wv & 3;
    const int wn   = wv >> 2;
    const int mBlock = blockIdx.x * BM;
    const int nBlock = blockIdx.y * BN;
    const int hh = lane >> 4;
    const int ml = lane & 15;

    const int arow = tid >> 1;
    const int acb  = (tid & 1) * 16;
    const int bnl  = tid >> 2;
    const int bkb  = (tid & 3) * 8;
    const size_t aRowBase = (size_t)(mBlock + arow) * lda + acb;
    const int    bValid   = (nBlock + bnl) < Nt;
    const size_t bRowBase = (size_t)(nBlock + bnl) * ldw + bkb;

    v8f acc[2][2];
    for (int i = 0; i < 2; ++i)
        for (int j = 0; j < 2; ++j)
            for (int e = 0; e < 8; ++e) acc[i][j][e] = 0.f;

    for (int k0 = 0; k0 < K; k0 += KC) {
        {
            h8v a0 = *(const h8v*)&A[aRowBase + k0];
            h8v a1 = *(const h8v*)&A[aRowBase + k0 + 8];
            *(h8v*)&As[arow * LDA_S + acb]     = a0;
            *(h8v*)&As[arow * LDA_S + acb + 8] = a1;
            if (k0 + KC < K)
                __builtin_prefetch(&A[aRowBase + k0 + KC], 0, 1); // global_prefetch_b8
        }
        {
            h8v w8;
            if (bValid) w8 = *(const h8v*)&W[bRowBase + k0];
            else for (int c = 0; c < 8; ++c) w8[c] = (_Float16)0.f;
            for (int c = 0; c < 8; ++c)
                Bs[(bkb + c) * LDB_S + bnl] = w8[c];
        }
        __syncthreads();

        v16h afr[2], bfr[2];
        for (int sa = 0; sa < 2; ++sa) {
            const int mrow = 32 * wm + 16 * sa + ml;
            const _Float16* ar = &As[mrow * LDA_S];
            for (int i = 0; i < 8; ++i) {
                const int kb = (i < 4 ? 2 * i : 16 + 2 * (i - 4)) + 8 * hh;
                afr[sa][2 * i]     = ar[kb];
                afr[sa][2 * i + 1] = ar[kb + 1];
            }
        }
        for (int sb = 0; sb < 2; ++sb) {
            const int nb = 32 * wn + 16 * sb;
            const _Float16* br = &Bs[nb + ml];
            for (int j = 0; j < 8; ++j) {
                const int kb = (j < 4 ? 2 * j : 16 + 2 * (j - 4)) + 8 * hh;
                bfr[sb][2 * j]     = br[kb * LDB_S];
                bfr[sb][2 * j + 1] = br[(kb + 1) * LDB_S];
            }
        }
        for (int sa = 0; sa < 2; ++sa)
            for (int sb = 0; sb < 2; ++sb)
                acc[sa][sb] = __builtin_amdgcn_wmma_f32_16x16x32_f16(
                    false, afr[sa], false, bfr[sb], (short)0, acc[sa][sb], false, false);
        __syncthreads();
    }

    for (int sa = 0; sa < 2; ++sa)
        for (int sb = 0; sb < 2; ++sb) {
            const int n = nBlock + 32 * wn + 16 * sb + ml;
            if (n >= Nt) continue;
            const float bn = bias ? bias[n] : 0.f;
            for (int r = 0; r < 8; ++r) {
                const int m = mBlock + 32 * wm + 16 * sa + r + 8 * hh;
                float f = acc[sa][sb][r] + bn;
                if (resid) f += resid[(size_t)m * ldo + n];
                if (relu)  f = fmaxf(f, 0.f);
                if (outF)  outF[(size_t)m * ldo + n] = f;
                if (outH)  outH[(size_t)m * ldo + n] = (_Float16)f;
            }
        }
}

// ---------------------------------------------------------------------------
// Specialized WMMA GEMM for K = lda = ldw = Nt = ldo = 128 (the dominant
// projections). The whole 128x128 fp16 A-tile is DMA'd into LDS ONCE per
// block by the Tensor Data Mover (tensor_load_to_lds, D# per ISA 08 §8),
// using TDM LDS padding (4 DWORDs every 64 DWORDs -> 136-half row stride)
// to keep fragment reads bank-conflict free. Wave 0 issues the DMA and
// drains TENSORcnt; a workgroup barrier publishes the tile.
// ---------------------------------------------------------------------------
__global__ __launch_bounds__(256) void gemm128_tdm_wmma(
    const _Float16* __restrict__ A,
    const _Float16* __restrict__ W,
    const float* __restrict__ bias,
    const float* __restrict__ resid,
    float* __restrict__ outF,
    _Float16* __restrict__ outH,
    int M, int relu)
{
    // Af is declared first -> LDS offset 0 (used as D#.lds_addr)
    __shared__ __align__(16) _Float16 Af[BM * AF_S];
    __shared__ __align__(16) _Float16 Bs[KC * LDB_S];

    const int tid  = threadIdx.x;
    const int lane = tid & 31;
    const int wv   = tid >> 5;
    const int wm   = wv & 3;
    const int wn   = wv >> 2;
    const int mBlock = blockIdx.x * BM;
    const int nBlock = blockIdx.y * BN;
    const int hh = lane >> 4;
    const int ml = lane & 15;

#if HAVE_TDM
    if (tid < 32) {
        const unsigned long long ga =
            (unsigned long long)(const void*)(A + (size_t)mBlock * DIMC);
        u32x4 g0; i32x8 g1; i32x4 g2, g3; i32x8 g4;
        // D# group0: count=1 | lds_addr | global_addr[56:0] | type=2
        g0[0] = 1u;
        g0[1] = 0u;                                     // Af at LDS offset 0
        g0[2] = (unsigned)(ga & 0xffffffffu);
        g0[3] = (unsigned)((ga >> 32) & 0x01ffffffu) | (2u << 30);
        // D# group1: data_size=2B | pad_enable, interval=64dw(code5), amount=4dw(code3)
        g1[0] = (int)((1u << 16) | (1u << 20) | (5u << 22) | (3u << 25));
        g1[1] = (int)(128u << 16);        // tensor_dim0[15:0]=128 (bits 79:48)
        g1[2] = (int)(128u << 16);        // tensor_dim0 hi=0 | tensor_dim1[15:0]=128
        g1[3] = (int)(128u << 16);        // tensor_dim1 hi=0 | tile_dim0=128
        g1[4] = (int)(128u);              // tile_dim1=128 rows | tile_dim2=0
        g1[5] = (int)128;                 // tensor_dim0_stride = 128 elements
        g1[6] = 0;                        // stride hi | tensor_dim1_stride lo
        g1[7] = 0;                        // tensor_dim1_stride hi
        g2[0] = g2[1] = g2[2] = g2[3] = 0;
        g3[0] = g3[1] = g3[2] = g3[3] = 0;
        for (int e = 0; e < 8; ++e) g4[e] = 0;
        // 6-arg form (clang-23 / therock-10.0 headers on this toolchain)
        __builtin_amdgcn_tensor_load_to_lds(g0, g1, g2, g3, g4, 0);
        __builtin_amdgcn_s_wait_tensorcnt(0);           // s_wait_tensorcnt 0
    }
    __syncthreads();
#else
    // cooperative fallback: 64 halfs per thread into the padded tile
    {
        const int r  = tid >> 1;
        const int cb = (tid & 1) * 64;
        for (int c = 0; c < 64; c += 8)
            *(h8v*)&Af[r * AF_S + cb + c] =
                *(const h8v*)&A[(size_t)(mBlock + r) * DIMC + cb + c];
    }
    __syncthreads();
#endif

    const int bnl = tid >> 2;
    const int bkb = (tid & 3) * 8;
    const size_t bRowBase = (size_t)(nBlock + bnl) * DIMC + bkb;

    v8f acc[2][2];
    for (int i = 0; i < 2; ++i)
        for (int j = 0; j < 2; ++j)
            for (int e = 0; e < 8; ++e) acc[i][j][e] = 0.f;

    for (int k0 = 0; k0 < DIMC; k0 += KC) {
        // stage B chunk transposed (Nt=128 -> always valid)
        {
            h8v w8 = *(const h8v*)&W[bRowBase + k0];
            for (int c = 0; c < 8; ++c)
                Bs[(bkb + c) * LDB_S + bnl] = w8[c];
        }
        __syncthreads();

        v16h afr[2], bfr[2];
        for (int sa = 0; sa < 2; ++sa) {
            const int mrow = 32 * wm + 16 * sa + ml;
            const _Float16* ar = &Af[mrow * AF_S + k0];
            for (int i = 0; i < 8; ++i) {
                const int kb = (i < 4 ? 2 * i : 16 + 2 * (i - 4)) + 8 * hh;
                afr[sa][2 * i]     = ar[kb];
                afr[sa][2 * i + 1] = ar[kb + 1];
            }
        }
        for (int sb = 0; sb < 2; ++sb) {
            const int nb = 32 * wn + 16 * sb;
            const _Float16* br = &Bs[nb + ml];
            for (int j = 0; j < 8; ++j) {
                const int kb = (j < 4 ? 2 * j : 16 + 2 * (j - 4)) + 8 * hh;
                bfr[sb][2 * j]     = br[kb * LDB_S];
                bfr[sb][2 * j + 1] = br[(kb + 1) * LDB_S];
            }
        }
        for (int sa = 0; sa < 2; ++sa)
            for (int sb = 0; sb < 2; ++sb)
                acc[sa][sb] = __builtin_amdgcn_wmma_f32_16x16x32_f16(
                    false, afr[sa], false, bfr[sb], (short)0, acc[sa][sb], false, false);
        __syncthreads();
    }

    for (int sa = 0; sa < 2; ++sa)
        for (int sb = 0; sb < 2; ++sb) {
            const int n = nBlock + 32 * wn + 16 * sb + ml;
            const float bn = bias ? bias[n] : 0.f;
            for (int r = 0; r < 8; ++r) {
                const int m = mBlock + 32 * wm + 16 * sa + r + 8 * hh;
                float f = acc[sa][sb][r] + bn;
                if (resid) f += resid[(size_t)m * DIMC + n];
                if (relu)  f = fmaxf(f, 0.f);
                if (outF)  outF[(size_t)m * DIMC + n] = f;
                if (outH)  outH[(size_t)m * DIMC + n] = (_Float16)f;
            }
        }
}

// ---------------------------------------------------------------------------
// Fused per-(ray, head) self-attention over S=64, dh=32.
// ---------------------------------------------------------------------------
__global__ __launch_bounds__(128) void self_attn_wmma(
    const _Float16* __restrict__ Q, const _Float16* __restrict__ Km,
    const _Float16* __restrict__ Vm, _Float16* __restrict__ Out)
{
    __shared__ __align__(16) _Float16 probs[4 * 16 * 64];

    const int blk  = blockIdx.x;
    const int n    = blk >> 2;
    const int head = blk & 3;
    const int hcol = head * DH;
    const int wv   = threadIdx.x >> 5;
    const int lane = threadIdx.x & 31;
    const int ml   = lane & 15;
    const int hh   = lane >> 4;
    const int s0   = wv * 16;
    const size_t rowbase = (size_t)n * SS;

    v16h aq;
    {
        const size_t r = (rowbase + s0 + ml) * DIMC + hcol;
        for (int i = 0; i < 8; ++i) {
            const int kb = (i < 4 ? 2 * i : 16 + 2 * (i - 4)) + 8 * hh;
            h2v t = *(const h2v*)&Q[r + kb];
            aq[2 * i] = t[0]; aq[2 * i + 1] = t[1];
        }
    }

    v8f sc[4];
    for (int t = 0; t < 4; ++t) {
        v16h bk;
        for (int j = 0; j < 8; ++j) {
            const int kb = (j < 4 ? 2 * j : 16 + 2 * (j - 4)) + 8 * hh;
            h2v tk = *(const h2v*)&Km[(rowbase + 16 * t + ml) * DIMC + hcol + kb];
            bk[2 * j]     = tk[0];
            bk[2 * j + 1] = tk[1];
        }
        v8f z; for (int e = 0; e < 8; ++e) z[e] = 0.f;
        sc[t] = __builtin_amdgcn_wmma_f32_16x16x32_f16(
            false, aq, false, bk, (short)0, z, false, false);
    }

    const float scale = 0.17677669529663687f; // 1/sqrt(32)
    for (int r = 0; r < 8; ++r) {
        float mx = -1e30f;
        for (int t = 0; t < 4; ++t) { sc[t][r] *= scale; mx = fmaxf(mx, sc[t][r]); }
        for (int m2 = 8; m2 >= 1; m2 >>= 1) mx = fmaxf(mx, __shfl_xor(mx, m2, 32));
        float s = 0.f;
        for (int t = 0; t < 4; ++t) { float e = __expf(sc[t][r] - mx); sc[t][r] = e; s += e; }
        for (int m2 = 8; m2 >= 1; m2 >>= 1) s += __shfl_xor(s, m2, 32);
        const float inv = 1.f / s;
        for (int t = 0; t < 4; ++t) sc[t][r] *= inv;
    }

    for (int t = 0; t < 4; ++t)
        for (int r = 0; r < 8; ++r)
            probs[(wv * 16 + r + 8 * hh) * 64 + 16 * t + ml] = (_Float16)sc[t][r];
    __syncthreads();

    v8f oacc[2];
    for (int t = 0; t < 2; ++t) for (int e = 0; e < 8; ++e) oacc[t][e] = 0.f;
    for (int c = 0; c < 2; ++c) {
        v16h ap;
        const _Float16* pr = &probs[(wv * 16 + ml) * 64 + 32 * c];
        for (int i = 0; i < 8; ++i) {
            const int kb = (i < 4 ? 2 * i : 16 + 2 * (i - 4)) + 8 * hh;
            ap[2 * i] = pr[kb]; ap[2 * i + 1] = pr[kb + 1];
        }
        for (int t = 0; t < 2; ++t) {
            v16h bv;
            const size_t rr = (rowbase + 32 * c) * DIMC + hcol + 16 * t + ml;
            for (int j = 0; j < 8; ++j) {
                const int kb = (j < 4 ? 2 * j : 16 + 2 * (j - 4)) + 8 * hh;
                bv[2 * j] = Vm[rr + (size_t)kb * DIMC]; bv[2 * j + 1] = Vm[rr + (size_t)(kb + 1) * DIMC];
            }
            oacc[t] = __builtin_amdgcn_wmma_f32_16x16x32_f16(
                false, ap, false, bv, (short)0, oacc[t], false, false);
        }
    }
    for (int t = 0; t < 2; ++t)
        for (int r = 0; r < 8; ++r) {
            const int s = s0 + r + 8 * hh;
            const int d = hcol + 16 * t + ml;
            Out[(rowbase + s) * DIMC + d] = (_Float16)oacc[t][r];
        }
}

// ---------------------------------------------------------------------------
// Elementwise / reduction helpers
// ---------------------------------------------------------------------------
__global__ void f32_to_f16_k(const float* __restrict__ src, _Float16* __restrict__ dst, long n)
{
    long i = (long)blockIdx.x * blockDim.x + threadIdx.x;
    if (i < n) dst[i] = (_Float16)src[i];
}

__global__ void zero_f16_k(_Float16* __restrict__ p, long n)
{
    long i = (long)blockIdx.x * blockDim.x + threadIdx.x;
    if (i < n) p[i] = (_Float16)0.f;
}

__global__ void convert_pad_k(const float* __restrict__ src, _Float16* __restrict__ dst,
                              long rows, int K, int Kpad)
{
    long idx = (long)blockIdx.x * blockDim.x + threadIdx.x;
    if (idx >= rows * (long)Kpad) return;
    const long r = idx / Kpad; const int c = (int)(idx - r * Kpad);
    dst[idx] = (c < K) ? (_Float16)src[r * (long)K + c] : (_Float16)0.f;
}

__global__ void embed_build_k(const float* __restrict__ pts, const float* __restrict__ ray_d,
                              _Float16* __restrict__ emb)
{
    int idx = blockIdx.x * blockDim.x + threadIdx.x;
    if (idx >= NS) return;
    const int n = idx / SS;
    float p[3], vd[3];
    p[0] = pts[(size_t)idx * 3]; p[1] = pts[(size_t)idx * 3 + 1]; p[2] = pts[(size_t)idx * 3 + 2];
    float r0 = ray_d[(size_t)n * 3], r1 = ray_d[(size_t)n * 3 + 1], r2 = ray_d[(size_t)n * 3 + 2];
    const float rn = rsqrtf(r0 * r0 + r1 * r1 + r2 * r2);
    vd[0] = r0 * rn; vd[1] = r1 * rn; vd[2] = r2 * rn;
    _Float16* o = emb + (size_t)idx * DIMC;
    for (int c = 0; c < 3; ++c) { o[c] = (_Float16)p[c]; o[63 + c] = (_Float16)vd[c]; }
    for (int f = 0; f < 10; ++f) {
        const float fr = exp2f((float)f);
        for (int c = 0; c < 3; ++c) {
            o[3 + f * 6 + c]      = (_Float16)sinf(fr * p[c]);
            o[3 + f * 6 + 3 + c]  = (_Float16)cosf(fr * p[c]);
            o[66 + f * 6 + c]     = (_Float16)sinf(fr * vd[c]);
            o[66 + f * 6 + 3 + c] = (_Float16)cosf(fr * vd[c]);
        }
    }
    o[126] = (_Float16)0.f; o[127] = (_Float16)0.f;
}

__global__ void maxv_k(const _Float16* __restrict__ feat, float* __restrict__ q32,
                       _Float16* __restrict__ q16)
{
    long idx = (long)blockIdx.x * blockDim.x + threadIdx.x;
    if (idx >= (long)NS * DIMC) return;
    const long ns = idx >> 7; const int d = idx & 127;
    float m = -1e30f;
    for (int v = 0; v < VV; ++v)
        m = fmaxf(m, (float)feat[((size_t)ns * VV + v) * DIMC + d]);
    q32[idx] = m; q16[idx] = (_Float16)m;
}

__global__ void layernorm_k(const float* __restrict__ X, const float* __restrict__ g,
                            const float* __restrict__ b, _Float16* __restrict__ oH,
                            float* __restrict__ oF, int M)
{
    const int lane = threadIdx.x & 31;
    const int wv   = threadIdx.x >> 5;
    const int row  = blockIdx.x * 8 + wv;
    if (row >= M) return;
    const float* x = X + (size_t)row * DIMC;
    float v[4]; float s = 0.f;
    for (int i = 0; i < 4; ++i) { v[i] = x[lane * 4 + i]; s += v[i]; }
    for (int m2 = 16; m2 >= 1; m2 >>= 1) s += __shfl_xor(s, m2, 32);
    const float mean = s * (1.f / 128.f);
    float vs = 0.f;
    for (int i = 0; i < 4; ++i) { const float d = v[i] - mean; vs += d * d; }
    for (int m2 = 16; m2 >= 1; m2 >>= 1) vs += __shfl_xor(vs, m2, 32);
    const float rstd = rsqrtf(vs * (1.f / 128.f) + 1e-6f);
    for (int i = 0; i < 4; ++i) {
        const int c = lane * 4 + i;
        const float y = (v[i] - mean) * rstd * g[c] + b[c];
        if (oH) oH[(size_t)row * DIMC + c] = (_Float16)y;
        if (oF) oF[(size_t)row * DIMC + c] = y;
    }
}

__global__ void a_build_k(const _Float16* __restrict__ kb, const _Float16* __restrict__ qe,
                          const _Float16* __restrict__ pos, _Float16* __restrict__ a)
{
    long idx = (long)blockIdx.x * blockDim.x + threadIdx.x;
    if (idx >= (long)NSV * DIMC) return;
    const long nsv = idx >> 7; const int d = idx & 127;
    const long ns = nsv / VV;
    a[idx] = (_Float16)((float)kb[idx] - (float)qe[ns * DIMC + d] + (float)pos[idx]);
}

__global__ void softmax_v_k(const _Float16* __restrict__ logits, const int* __restrict__ mask,
                            const _Float16* __restrict__ vb, const _Float16* __restrict__ pos,
                            _Float16* __restrict__ out)
{
    long idx = (long)blockIdx.x * blockDim.x + threadIdx.x;
    if (idx >= (long)NS * DIMC) return;
    const long ns = idx >> 7; const int d = idx & 127;
    float lg[VV]; float mx = -1e30f;
    for (int v = 0; v < VV; ++v) {
        const long mb = ns * VV + v;
        float l = (float)logits[mb * DIMC + d];
        if (mask[mb] == 0) l = -1e9f;
        lg[v] = l; mx = fmaxf(mx, l);
    }
    float s = 0.f;
    for (int v = 0; v < VV; ++v) { lg[v] = __expf(lg[v] - mx); s += lg[v]; }
    const float inv = 1.f / s;
    float acc = 0.f;
    for (int v = 0; v < VV; ++v) {
        const long mb = (ns * VV + v) * DIMC + d;
        acc += ((float)vb[mb] + (float)pos[mb]) * lg[v] * inv;
    }
    out[idx] = (_Float16)acc;
}

__global__ void qc_build_k(const _Float16* __restrict__ q16, const _Float16* __restrict__ emb,
                           _Float16* __restrict__ qc)
{
    long idx = (long)blockIdx.x * blockDim.x + threadIdx.x;
    if (idx >= (long)NS * 256) return;
    const long ns = idx >> 8; const int c = idx & 255;
    _Float16 v = (_Float16)0.f;
    if (c < 128) v = q16[ns * DIMC + c];
    else if (c < 254) v = emb[ns * DIMC + (c - 128)];
    qc[idx] = v;
}

__global__ void final_out_k(const float* __restrict__ hfin, const float* __restrict__ w,
                            const float* __restrict__ b, float* __restrict__ out)
{
    __shared__ float mv[DIMC];
    const int n = blockIdx.x; const int d = threadIdx.x;
    float s = 0.f;
    for (int si = 0; si < SS; ++si) s += hfin[((size_t)n * SS + si) * DIMC + d];
    mv[d] = s * (1.f / (float)SS);
    __syncthreads();
    if (d < 3) {
        float acc = b[d];
        for (int c = 0; c < DIMC; ++c) acc += mv[c] * w[d * DIMC + c];
        out[(size_t)n * 3 + d] = acc;
    }
}

// ---------------------------------------------------------------------------
// Host driver
// ---------------------------------------------------------------------------
extern "C" void kernel_launch(void* const* d_in, const int* in_sizes, int n_in,
                              void* d_out, int out_size, void* d_ws, size_t ws_size,
                              hipStream_t stream)
{
    const float* rgb_feat = (const float*)d_in[0];
    const float* ray_diff = (const float*)d_in[1];
    const int*   mask     = (const int*)d_in[2];
    const float* pts      = (const float*)d_in[3];
    const float* ray_d    = (const float*)d_in[4];
    const float* Pf       = (const float*)d_in[7];
    float* outp           = (float*)d_out;

    struct LOff {
        size_t an_b, an_g, attn1_b, attn1_w, attn2_b, attn2_w, embed_;
        size_t ff1_b, ff1_w, ff2_b, ff2_w, fn_b, fn_g, k_w, out_b, out_w;
        size_t pos1_b, pos1_w, pos2_b, pos2_w, v_w;
        size_t q1_b, q1_w, q2_b, q2_w;
        size_t s_an_b, s_an_g, s_embed, s_ff1_b, s_ff1_w, s_ff2_b, s_ff2_w;
        size_t s_fn_b, s_fn_g, s_k_w, s_out_b, s_out_w, s_q_w;
    };
    LOff Ls[8] = {};
    size_t off = 0;
    auto take = [&](size_t nEl) { size_t o = off; off += nEl; return o; };
    for (int i = 0; i < 8; ++i) {
        Ls[i].an_b = take(128);  Ls[i].an_g = take(128);
        Ls[i].attn1_b = take(16);  Ls[i].attn1_w = take(16 * 128);
        Ls[i].attn2_b = take(128); Ls[i].attn2_w = take(128 * 16);
        Ls[i].embed_ = take(128 * 128);
        Ls[i].ff1_b = take(512); Ls[i].ff1_w = take(512 * 128);
        Ls[i].ff2_b = take(128); Ls[i].ff2_w = take(128 * 512);
        Ls[i].fn_b = take(128);  Ls[i].fn_g = take(128);
        Ls[i].k_w = take(128 * 128);
        Ls[i].out_b = take(128); Ls[i].out_w = take(128 * 128);
        Ls[i].pos1_b = take(16);  Ls[i].pos1_w = take(16 * 4);
        Ls[i].pos2_b = take(128); Ls[i].pos2_w = take(128 * 16);
        Ls[i].v_w = take(128 * 128);
        if (i % 2 == 0) {
            Ls[i].q1_b = take(128); Ls[i].q1_w = take(128 * 254);
            Ls[i].q2_b = take(128); Ls[i].q2_w = take(128 * 128);
        }
        Ls[i].s_an_b = take(128); Ls[i].s_an_g = take(128);
        Ls[i].s_embed = take(128 * 128);
        Ls[i].s_ff1_b = take(512); Ls[i].s_ff1_w = take(512 * 128);
        Ls[i].s_ff2_b = take(128); Ls[i].s_ff2_w = take(128 * 512);
        Ls[i].s_fn_b = take(128); Ls[i].s_fn_g = take(128);
        Ls[i].s_k_w = take(128 * 128);
        Ls[i].s_out_b = take(128); Ls[i].s_out_w = take(128 * 128);
        Ls[i].s_q_w = take(128 * 128);
    }
    const size_t o_norm_b  = take(128);
    const size_t o_norm_g  = take(128);
    const size_t o_rgb1_b  = take(128);
    const size_t o_rgb1_w  = take(128 * 35);
    const size_t o_rgb2_b  = take(128);
    const size_t o_rgb2_w  = take(128 * 128);
    const size_t o_rgbfc_b = take(3);
    const size_t o_rgbfc_w = take(3 * 128);
    const size_t nparam = off;

    char* base = (char*)d_ws;
    size_t cur = 0;
    auto alloc = [&](size_t bytes) -> void* {
        void* p = base + cur;
        cur = (cur + bytes + 255) & ~(size_t)255;
        return p;
    };
    _Float16* P16      = (_Float16*)alloc(nparam * 2);
    _Float16* rf16     = (_Float16*)alloc((size_t)NSV * 64 * 2);
    _Float16* rd16     = (_Float16*)alloc((size_t)NSV * 32 * 2);
    _Float16* emb16    = (_Float16*)alloc((size_t)NS * DIMC * 2);
    _Float16* feat16   = (_Float16*)alloc((size_t)NSV * DIMC * 2);
    float*    q32      = (float*)alloc((size_t)NS * DIMC * 4);
    _Float16* q16      = (_Float16*)alloc((size_t)NS * DIMC * 2);
    _Float16* xln16    = (_Float16*)alloc((size_t)NS * DIMC * 2);
    _Float16* qe16     = (_Float16*)alloc((size_t)NS * DIMC * 2);
    _Float16* t16a     = (_Float16*)alloc((size_t)NS * DIMC * 2);
    _Float16* sattn16  = (_Float16*)alloc((size_t)NS * DIMC * 2);
    _Float16* qc16     = (_Float16*)alloc((size_t)NS * 256 * 2);
    _Float16* hid16    = (_Float16*)alloc((size_t)NS * 512 * 2);
    float*    x32      = (float*)alloc((size_t)NS * DIMC * 4);
    _Float16* qp16     = (_Float16*)alloc((size_t)NS * DIMC * 2);
    _Float16* kp16     = (_Float16*)alloc((size_t)NS * DIMC * 2);
    _Float16* vp16     = (_Float16*)alloc((size_t)NS * DIMC * 2);
    _Float16* kbuf16   = (_Float16*)alloc((size_t)NSV * DIMC * 2);
    _Float16* vbuf16   = (_Float16*)alloc((size_t)NSV * DIMC * 2);
    _Float16* posbuf16 = (_Float16*)alloc((size_t)NSV * DIMC * 2);
    _Float16* abuf16   = (_Float16*)alloc((size_t)NSV * DIMC * 2);
    _Float16* poshid16 = (_Float16*)alloc((size_t)NSV * 32 * 2);
    _Float16* rgb1p = (_Float16*)alloc((size_t)128 * 64 * 2);
    _Float16* pos1p = (_Float16*)alloc((size_t)8 * 16 * 32 * 2);
    _Float16* pos2p = (_Float16*)alloc((size_t)8 * 128 * 32 * 2);
    _Float16* atn2p = (_Float16*)alloc((size_t)8 * 128 * 32 * 2);
    _Float16* q1p   = (_Float16*)alloc((size_t)4 * 128 * 256 * 2);
    (void)ws_size; (void)in_sizes; (void)n_in; (void)out_size;

    auto EW = [&](long n) { return dim3((unsigned)((n + 255) / 256)); };
    auto GEMM = [&](const _Float16* Ain, int lda, const _Float16* Wp, int ldw,
                    const float* biasPtr, const float* residPtr, float* oF, _Float16* oH,
                    int M, int Nt, int K, int ldo, int relu) {
        if (lda == DIMC && ldw == DIMC && K == DIMC && Nt == DIMC && ldo == DIMC) {
            dim3 grid(M / BM, DIMC / BN);
            gemm128_tdm_wmma<<<grid, 256, 0, stream>>>(Ain, Wp, biasPtr, residPtr,
                                                       oF, oH, M, relu);
        } else {
            dim3 grid((M + BM - 1) / BM, (Nt + BN - 1) / BN);
            gemm_f16_wmma<<<grid, 256, 0, stream>>>(Ain, lda, Wp, ldw, biasPtr,
                                                    residPtr, oF, oH, M, Nt, K, ldo, relu);
        }
    };
    auto PACKW = [&](size_t srcOff, _Float16* dst, int Nt, int K, int Kpad) {
        convert_pad_k<<<EW((long)Nt * Kpad), 256, 0, stream>>>(Pf + srcOff, dst, Nt, K, Kpad);
    };
    auto LNRM = [&](const float* X, size_t gOff, size_t bOff, _Float16* oH, float* oF) {
        layernorm_k<<<(NS + 7) / 8, 256, 0, stream>>>(X, Pf + gOff, Pf + bOff, oH, oF, NS);
    };

    f32_to_f16_k<<<EW((long)nparam), 256, 0, stream>>>(Pf, P16, (long)nparam);
    convert_pad_k<<<EW((long)NSV * 64), 256, 0, stream>>>(rgb_feat, rf16, NSV, FEATC, 64);
    convert_pad_k<<<EW((long)NSV * 32), 256, 0, stream>>>(ray_diff, rd16, NSV, 4, 32);
    zero_f16_k<<<EW((long)NSV * 32), 256, 0, stream>>>(poshid16, (long)NSV * 32);
    embed_build_k<<<EW(NS), 256, 0, stream>>>(pts, ray_d, emb16);
    PACKW(o_rgb1_w, rgb1p, 128, 35, 64);
    for (int i = 0; i < 8; ++i) {
        PACKW(Ls[i].pos1_w, pos1p + (size_t)i * 16 * 32, 16, 4, 32);
        PACKW(Ls[i].pos2_w, pos2p + (size_t)i * 128 * 32, 128, 16, 32);
        PACKW(Ls[i].attn2_w, atn2p + (size_t)i * 128 * 32, 128, 16, 32);
        if (i % 2 == 0)
            PACKW(Ls[i].q1_w, q1p + (size_t)(i / 2) * 128 * 256, 128, 254, 256);
    }

    GEMM(rf16, 64, rgb1p, 64, Pf + o_rgb1_b, nullptr, nullptr, kbuf16, NSV, DIMC, 64, DIMC, 1);
    GEMM(kbuf16, DIMC, P16 + o_rgb2_w, DIMC, Pf + o_rgb2_b, nullptr, nullptr, feat16,
         NSV, DIMC, DIMC, DIMC, 0);
    maxv_k<<<EW((long)NS * DIMC), 256, 0, stream>>>(feat16, q32, q16);

    for (int i = 0; i < 8; ++i) {
        const LOff& L = Ls[i];
        // ======== view cross-transformer ========
        LNRM(q32, L.an_g, L.an_b, xln16, nullptr);
        GEMM(xln16, DIMC, P16 + L.embed_, DIMC, nullptr, nullptr, nullptr, qe16,
             NS, DIMC, DIMC, DIMC, 0);
        GEMM(feat16, DIMC, P16 + L.k_w, DIMC, nullptr, nullptr, nullptr, kbuf16,
             NSV, DIMC, DIMC, DIMC, 0);
        GEMM(kbuf16, DIMC, P16 + L.v_w, DIMC, nullptr, nullptr, nullptr, vbuf16,
             NSV, DIMC, DIMC, DIMC, 0);
        GEMM(rd16, 32, pos1p + (size_t)i * 16 * 32, 32, Pf + L.pos1_b, nullptr, nullptr,
             poshid16, NSV, 16, 32, 32, 1);
        GEMM(poshid16, 32, pos2p + (size_t)i * 128 * 32, 32, Pf + L.pos2_b, nullptr, nullptr,
             posbuf16, NSV, DIMC, 32, DIMC, 0);
        a_build_k<<<EW((long)NSV * DIMC), 256, 0, stream>>>(kbuf16, qe16, posbuf16, abuf16);
        GEMM(abuf16, DIMC, P16 + L.attn1_w, DIMC, Pf + L.attn1_b, nullptr, nullptr,
             poshid16, NSV, 16, DIMC, 32, 1);
        GEMM(poshid16, 32, atn2p + (size_t)i * 128 * 32, 32, Pf + L.attn2_b, nullptr, nullptr,
             abuf16, NSV, DIMC, 32, DIMC, 0);
        softmax_v_k<<<EW((long)NS * DIMC), 256, 0, stream>>>(abuf16, mask, vbuf16, posbuf16, t16a);
        GEMM(t16a, DIMC, P16 + L.out_w, DIMC, Pf + L.out_b, q32, x32, nullptr,
             NS, DIMC, DIMC, DIMC, 0);
        LNRM(x32, L.fn_g, L.fn_b, xln16, nullptr);
        GEMM(xln16, DIMC, P16 + L.ff1_w, DIMC, Pf + L.ff1_b, nullptr, nullptr, hid16,
             NS, 512, DIMC, 512, 1);
        GEMM(hid16, 512, P16 + L.ff2_w, 512, Pf + L.ff2_b, x32, q32, q16,
             NS, DIMC, 512, DIMC, 0);

        // ======== even layers: q-concat MLP ========
        if (i % 2 == 0) {
            qc_build_k<<<EW((long)NS * 256), 256, 0, stream>>>(q16, emb16, qc16);
            GEMM(qc16, 256, q1p + (size_t)(i / 2) * 128 * 256, 256, Pf + L.q1_b, nullptr,
                 nullptr, t16a, NS, DIMC, 256, DIMC, 1);
            GEMM(t16a, DIMC, P16 + L.q2_w, DIMC, Pf + L.q2_b, nullptr, q32, q16,
                 NS, DIMC, DIMC, DIMC, 0);
        }

        // ======== ray self-transformer ========
        LNRM(q32, L.s_an_g, L.s_an_b, xln16, nullptr);
        GEMM(xln16, DIMC, P16 + L.s_q_w, DIMC, nullptr, nullptr, nullptr, qp16,
             NS, DIMC, DIMC, DIMC, 0);
        GEMM(xln16, DIMC, P16 + L.s_k_w, DIMC, nullptr, nullptr, nullptr, kp16,
             NS, DIMC, DIMC, DIMC, 0);
        GEMM(xln16, DIMC, P16 + L.s_embed, DIMC, nullptr, nullptr, nullptr, vp16,
             NS, DIMC, DIMC, DIMC, 0);
        self_attn_wmma<<<NRAY * HEADS, 128, 0, stream>>>(qp16, kp16, vp16, sattn16);
        GEMM(sattn16, DIMC, P16 + L.s_out_w, DIMC, Pf + L.s_out_b, q32, x32, nullptr,
             NS, DIMC, DIMC, DIMC, 0);
        LNRM(x32, L.s_fn_g, L.s_fn_b, xln16, nullptr);
        GEMM(xln16, DIMC, P16 + L.s_ff1_w, DIMC, Pf + L.s_ff1_b, nullptr, nullptr, hid16,
             NS, 512, DIMC, 512, 1);
        GEMM(hid16, 512, P16 + L.s_ff2_w, 512, Pf + L.s_ff2_b, x32, q32, q16,
             NS, DIMC, 512, DIMC, 0);
    }

    LNRM(q32, o_norm_g, o_norm_b, nullptr, x32);
    final_out_k<<<NRAY, DIMC, 0, stream>>>(x32, Pf + o_rgbfc_w, Pf + o_rgbfc_b, outp);
}
